// DotAttentionUnit_57552561766573
// MI455X (gfx1250) — matrix operand order repaired and checked by reading, not compiled, not run on hardware
//
#include <hip/hip_runtime.h>

// ---------------------------------------------------------------------------
// DotAttentionUnit for MI455X (gfx1250), wave32 + WMMA.
//   s_pre = (hq*hp) @ W^T  via v_wmma_f32_16x16x32_f16, operands staged in the
//   320KB WGP LDS; tanh via hw v_tanh_f32; softmax + a@hq fused in-kernel.
// One workgroup per (b,p): grid = (LP, B), 256 threads (8 waves).
// Round 3: software pipeline with peeled epilogue + fully unrolled K loop so
// every ds_load keeps an immediate offset (no address VALU -> no WMMA hazard
// NOPs) while next-step loads stay in flight during the WMMAs.
// ---------------------------------------------------------------------------

typedef __attribute__((ext_vector_type(16))) _Float16 v16h;
typedef __attribute__((ext_vector_type(8)))  _Float16 v8h;
typedef __attribute__((ext_vector_type(8)))  float    v8f;
typedef __attribute__((ext_vector_type(4)))  float    v4f;

constexpr int B_  = 4;
constexpr int LQ_ = 256;
constexpr int LP_ = 256;
constexpr int H_  = 512;
constexpr int V_  = 512;

constexpr int NT    = 32;        // v-tile (N) columns staged per iteration
constexpr int KSTEP = 32;        // K per wmma_f32_16x16x32_f16
constexpr int XS    = H_ + 8;    // padded f16 row stride (520) -> bank-spread

constexpr size_t X_BYTES  = (size_t)LQ_ * XS * sizeof(_Float16);  // 266240
constexpr size_t WT_BYTES = (size_t)NT  * XS * sizeof(_Float16);  //  33280
constexpr size_t SM_BYTES = X_BYTES + WT_BYTES + 3 * 256 * sizeof(float); // 302592 < 320KB

union F16Frag { v16h v; v8h h[2]; };

#if __has_builtin(__builtin_amdgcn_tanhf)
__device__ __forceinline__ float fast_tanh(float x) { return __builtin_amdgcn_tanhf(x); }
#else
__device__ __forceinline__ float fast_tanh(float x) { return tanhf(x); }
#endif

// Load one k-step worth of fragments (2 A tiles, 2 B tiles) from LDS.
// A frag (16x32 f16): lanes 0-15 hold K[k..k+8)+[k+16..k+24),
//                     lanes 16-31 hold K[k+8..k+16)+[k+24..k+32).
// B frag (32x16 f16): lane n = column v; lanes 0-15 K[k..k+8)+[k+16..k+24),
//                     lanes 16-31 K[k+8..k+16)+[k+24..k+32).
__device__ __forceinline__ void load_frags(const _Float16* __restrict__ xrow0,
                                           const _Float16* __restrict__ xrow1,
                                           const _Float16* __restrict__ wrow,
                                           int k, int half,
                                           F16Frag& A0, F16Frag& A1,
                                           F16Frag& B0, F16Frag& B1)
{
    const int aoff = k + half * 8;
    A0.h[0] = *(const v8h*)(xrow0 + aoff);
    A0.h[1] = *(const v8h*)(xrow0 + aoff + 16);
    A1.h[0] = *(const v8h*)(xrow1 + aoff);
    A1.h[1] = *(const v8h*)(xrow1 + aoff + 16);
    const int boff = k + half * 8;
    B0.h[0] = *(const v8h*)(wrow + boff);
    B0.h[1] = *(const v8h*)(wrow + boff + 16);
    B1.h[0] = *(const v8h*)(wrow + (size_t)16 * XS + boff);
    B1.h[1] = *(const v8h*)(wrow + (size_t)16 * XS + boff + 16);
}

__global__ void __launch_bounds__(256)
DotAttentionUnit_57552561766573_kernel(const float* __restrict__ hq,
                                       const float* __restrict__ hp,
                                       const float* __restrict__ W,
                                       const float* __restrict__ vw,
                                       float* __restrict__ out)
{
    extern __shared__ char smem[];
    _Float16* X   = (_Float16*)smem;                       // 256 x 520 f16 (hq*hp)
    _Float16* Wt  = (_Float16*)(smem + X_BYTES);           //  32 x 520 f16 (W tile)
    float* s_buf  = (float*)(smem + X_BYTES + WT_BYTES);   // 256 scores
    float* red    = s_buf + 256;                           // 256 reduction scratch
    float* a_buf  = red + 256;                             // 256 softmax weights

    const int tid  = threadIdx.x;
    const int wave = tid >> 5;
    const int lane = tid & 31;
    const int half = lane >> 4;   // 0: lanes 0-15, 1: lanes 16-31
    const int l16  = lane & 15;

    const int p = blockIdx.x;
    const int b = blockIdx.y;

    const float* hq_b = hq + (size_t)b * LQ_ * H_;
    const float* hp_r = hp + ((size_t)b * LP_ + p) * H_;

    // ---- Phase 1: X[q][h] = f16( hq[b,q,h] * hp[b,p,h] ) into LDS ----------
    for (int idx = tid; idx < LQ_ * (H_ / 8); idx += 256) {
        const int q  = idx >> 6;           // H_/8 == 64 chunks per row
        const int hc = (idx & 63) << 3;
        const v4f* a4 = (const v4f*)(hq_b + (size_t)q * H_ + hc);
        const v4f* b4 = (const v4f*)(hp_r + hc);
        v4f a0 = a4[0], a1 = a4[1];
        v4f b0 = b4[0], b1 = b4[1];
        v8h o;
#pragma unroll
        for (int i = 0; i < 4; ++i) o[i]     = (_Float16)(a0[i] * b0[i]);
#pragma unroll
        for (int i = 0; i < 4; ++i) o[4 + i] = (_Float16)(a1[i] * b1[i]);
        *(v8h*)(X + (size_t)q * XS + hc) = o;
    }
    __syncthreads();

    // ---- Phase 2: s[q] = sum_v vw[v] * tanh( (X @ W^T)[q,v] ) --------------
    const int mbase = wave * 32;           // each wave owns 32 q rows
    float sp[2][8];
#pragma unroll
    for (int mt = 0; mt < 2; ++mt)
#pragma unroll
        for (int r = 0; r < 8; ++r) sp[mt][r] = 0.0f;

    const _Float16* xrow0 = X + (size_t)(mbase + l16) * XS;
    const _Float16* xrow1 = xrow0 + (size_t)16 * XS;
    const _Float16* wrow  = Wt + (size_t)l16 * XS;

    for (int vbase = 0; vbase < V_; vbase += NT) {
        // Stage W tile: f32 -> f16 into LDS (cooperative, coalesced)
        for (int idx = tid; idx < NT * (H_ / 8); idx += 256) {
            const int v  = idx >> 6;
            const int hc = (idx & 63) << 3;
            const v4f* w4 = (const v4f*)(W + (size_t)(vbase + v) * H_ + hc);
            v4f w0 = w4[0], w1 = w4[1];
            v8h o;
#pragma unroll
            for (int i = 0; i < 4; ++i) o[i]     = (_Float16)w0[i];
#pragma unroll
            for (int i = 0; i < 4; ++i) o[4 + i] = (_Float16)w1[i];
            *(v8h*)(Wt + (size_t)v * XS + hc) = o;
            if (vbase + NT < V_)  // global_prefetch for next tile
                __builtin_prefetch(W + (size_t)(vbase + NT + v) * H_ + hc, 0, 1);
        }
        __syncthreads();

        v8f acc[2][2] = {};

        // Software-pipelined, fully unrolled K loop with peeled epilogue:
        // loads for step k+32 issue before the WMMAs of step k, and every ds
        // offset is a compile-time immediate (no address VALU in the loop).
        F16Frag Ac0, Ac1, Bc0, Bc1, An0, An1, Bn0, Bn1;
        load_frags(xrow0, xrow1, wrow, 0, half, Ac0, Ac1, Bc0, Bc1);
#pragma unroll
        for (int k = 0; k < H_ - KSTEP; k += KSTEP) {
            load_frags(xrow0, xrow1, wrow, k + KSTEP, half, An0, An1, Bn0, Bn1);

            acc[0][0] = __builtin_amdgcn_wmma_f32_16x16x32_f16(false, Ac0.v, false, Bc0.v, (short)0, acc[0][0], false, false);
            acc[0][1] = __builtin_amdgcn_wmma_f32_16x16x32_f16(false, Ac0.v, false, Bc1.v, (short)0, acc[0][1], false, false);
            acc[1][0] = __builtin_amdgcn_wmma_f32_16x16x32_f16(false, Ac1.v, false, Bc0.v, (short)0, acc[1][0], false, false);
            acc[1][1] = __builtin_amdgcn_wmma_f32_16x16x32_f16(false, Ac1.v, false, Bc1.v, (short)0, acc[1][1], false, false);

            Ac0 = An0; Ac1 = An1; Bc0 = Bn0; Bc1 = Bn1;   // renamed away by unroll
        }
        // Epilogue: last k-step (k = H_-KSTEP)
        acc[0][0] = __builtin_amdgcn_wmma_f32_16x16x32_f16(false, Ac0.v, false, Bc0.v, (short)0, acc[0][0], false, false);
        acc[0][1] = __builtin_amdgcn_wmma_f32_16x16x32_f16(false, Ac0.v, false, Bc1.v, (short)0, acc[0][1], false, false);
        acc[1][0] = __builtin_amdgcn_wmma_f32_16x16x32_f16(false, Ac1.v, false, Bc0.v, (short)0, acc[1][0], false, false);
        acc[1][1] = __builtin_amdgcn_wmma_f32_16x16x32_f16(false, Ac1.v, false, Bc1.v, (short)0, acc[1][1], false, false);

        __syncthreads();   // Wt fully consumed; next iteration may restage

        // C layout: VGPR r -> row M = r + half*8, lane -> column N = l16.
        // v is a per-lane scalar => fold tanh + vw weighting into row partials.
#pragma unroll
        for (int ns = 0; ns < 2; ++ns) {
            const float vwv = vw[vbase + ns * 16 + l16];
#pragma unroll
            for (int mt = 0; mt < 2; ++mt)
#pragma unroll
                for (int r = 0; r < 8; ++r)
                    sp[mt][r] = fmaf(vwv, fast_tanh(acc[mt][ns][r]), sp[mt][r]);
        }
    }

    // ---- Phase 3: cross-lane reduce partials into s_buf[q] -----------------
    s_buf[tid] = 0.0f;
    __syncthreads();
#pragma unroll
    for (int mt = 0; mt < 2; ++mt)
#pragma unroll
        for (int r = 0; r < 8; ++r)
            atomicAdd(&s_buf[mbase + mt * 16 + half * 8 + r], sp[mt][r]);
    __syncthreads();

    // ---- Phase 4: softmax over q (tid == q) --------------------------------
    const float sv = s_buf[tid];
    red[tid] = sv;
    __syncthreads();
    for (int s = 128; s > 0; s >>= 1) {
        if (tid < s) red[tid] = fmaxf(red[tid], red[tid + s]);
        __syncthreads();
    }
    const float mx = red[0];
    __syncthreads();
    const float e = __expf(sv - mx);
    red[tid] = e;
    __syncthreads();
    for (int s = 128; s > 0; s >>= 1) {
        if (tid < s) red[tid] += red[tid + s];
        __syncthreads();
    }
    const float denom = red[0];
    __syncthreads();
    a_buf[tid] = e / denom;
    __syncthreads();

    // ---- Phase 5: q_att[b,p,h] = sum_q a[q] * hq[b,q,h]  (tiny GEMV) -------
    const size_t obase = ((size_t)b * LP_ + p) * H_;
    for (int h = tid; h < H_; h += 256) {
        const float* col = hq_b + h;
        float acc = 0.0f;
#pragma unroll 8
        for (int q = 0; q < LQ_; ++q)
            acc = fmaf(a_buf[q], col[(size_t)q * H_], acc);
        out[obase + h] = acc;
    }
}

extern "C" void kernel_launch(void* const* d_in, const int* in_sizes, int n_in,
                              void* d_out, int out_size, void* d_ws, size_t ws_size,
                              hipStream_t stream) {
    (void)in_sizes; (void)n_in; (void)d_ws; (void)ws_size; (void)out_size;
    const float* hq = (const float*)d_in[0];
    const float* hp = (const float*)d_in[1];
    const float* W  = (const float*)d_in[2];
    const float* vw = (const float*)d_in[3];
    float* out = (float*)d_out;

    // Opt in to >64KB dynamic LDS (CDNA5 WGP has 320KB).
    (void)hipFuncSetAttribute((const void*)DotAttentionUnit_57552561766573_kernel,
                              hipFuncAttributeMaxDynamicSharedMemorySize,
                              (int)SM_BYTES);

    dim3 grid(LP_, B_);
    DotAttentionUnit_57552561766573_kernel<<<grid, 256, SM_BYTES, stream>>>(hq, hp, W, vw, out);
}
